// GCN_47339129536599
// MI455X (gfx1250) — matrix operand order repaired and checked by reading, not compiled, not run on hardware
//
#include <hip/hip_runtime.h>
#include <hip/hip_bf16.h>

typedef __attribute__((ext_vector_type(16))) __bf16 v16bf;
typedef __attribute__((ext_vector_type(8)))  __bf16 v8bf;
typedef __attribute__((ext_vector_type(4)))  __bf16 v4bf;
typedef __attribute__((ext_vector_type(8)))  float  v8f;

// ---------------------------------------------------------------------------
// Degree / normalization helpers
// ---------------------------------------------------------------------------
__global__ void fill_ones_f32(float* __restrict__ p, int n) {
  int i = blockIdx.x * blockDim.x + threadIdx.x;
  if (i < n) p[i] = 1.0f;   // start at 1.0: folds in the self-loop edge
}

__global__ void deg_accum(const int* __restrict__ dst, float* __restrict__ deg, int E) {
  int i = blockIdx.x * blockDim.x + threadIdx.x;
  if (i < E) atomicAdd(&deg[dst[i]], 1.0f);
}

__global__ void rsqrt_inplace(float* __restrict__ p, int n) {
  int i = blockIdx.x * blockDim.x + threadIdx.x;
  if (i < n) p[i] = rsqrtf(p[i]);   // deg >= 1 always (self-loop), no zero guard
}

// ---------------------------------------------------------------------------
// Activation pack: fp32 row-major -> bf16 row-major, ReLU fused at compile
// time. One pass per layer input; shared by all column-tile waves of the
// GEMM, so the f32->bf16 conversion happens once per row instead of NT x.
// ---------------------------------------------------------------------------
template <bool RELU>
__global__ void pack_a_bf16(const float* __restrict__ in, __bf16* __restrict__ out,
                            long total4) {
  long idx = (long)blockIdx.x * blockDim.x + threadIdx.x;
  if (idx >= total4) return;
  float4 f = ((const float4*)in)[idx];
  if (RELU) {
    f.x = fmaxf(f.x, 0.0f);
    f.y = fmaxf(f.y, 0.0f);
    f.z = fmaxf(f.z, 0.0f);
    f.w = fmaxf(f.w, 0.0f);
  }
  v4bf o;
  o[0] = (__bf16)f.x;
  o[1] = (__bf16)f.y;
  o[2] = (__bf16)f.z;
  o[3] = (__bf16)f.w;
  ((v4bf*)out)[idx] = o;
}

// ---------------------------------------------------------------------------
// Pack W [K x N] (f32 row-major) into per-lane wave32 WMMA B fragments (bf16).
// V_WMMA_F32_16X16X32_BF16 B-matrix (32x16) layout per ISA 7.12.2:
//   lane L: n = (L & 15); element i -> B[32*kt + (i>>3)*16 + (L>>4)*8 + (i&7)][n]
// Fragment for (kt, ct, lane) stored contiguously: 16 bf16 = one 32B load.
// ---------------------------------------------------------------------------
__global__ void pack_w_bf16(const float* __restrict__ W, __bf16* __restrict__ Wp,
                            int K, int N) {
  int idx = blockIdx.x * blockDim.x + threadIdx.x;
  int NT = N >> 4;
  int total = (K >> 5) * NT * 512;
  if (idx >= total) return;
  int i    = idx & 15;
  int lane = (idx >> 4) & 31;
  int t    = idx >> 9;                      // t = kt*NT + ct
  int ct   = t % NT;
  int kt   = t / NT;
  int nn   = (ct << 4) + (lane & 15);
  int k    = (kt << 5) + ((i >> 3) << 4) + ((lane >> 4) << 3) + (i & 7);
  Wp[idx] = (__bf16)W[k * N + nn];
}

// ---------------------------------------------------------------------------
// GEMM + fused GCN epilogue:
//   H[r][c]   = (Abf @ W)[r][c]                       (raw features, fp32)
//   Agg[r][c] = bias[c] + dinv[r]^2 * H[r][c]         (bias + self-loop edge)
// bf16 WMMA, f32 accumulate. One wave per 16x16 output tile; wave w owns
// column tile w, block b owns row tile b. K = 128 -> 4 chained
// v_wmma_f32_16x16x32_bf16. Inner loop per K-step: 2x 16B A loads (register
// concat) + 1x 32B B load (pre-packed register image) + 1 WMMA.
// ---------------------------------------------------------------------------
template <int NT>
__global__ void __launch_bounds__(32 * NT)
gemm_bf16_wmma(const __bf16* __restrict__ A, const __bf16* __restrict__ Wp,
               float* __restrict__ H, float* __restrict__ Agg,
               const float* __restrict__ dinv, const float* __restrict__ bias) {
  constexpr int K = 128;
  constexpr int KT = 4;
  constexpr int N = NT * 16;
  const int lane  = threadIdx.x & 31;
  const int ct    = threadIdx.x >> 5;          // column tile (wave id)
  const int m     = lane & 15;
  const int khalf = lane >> 4;                 // A half: 0 -> K{0..7,16..23}, 1 -> {8..15,24..31}
  const __bf16* __restrict__ arow  = A + ((long)blockIdx.x * 16 + m) * K + khalf * 8;
  const __bf16* __restrict__ bbase = Wp + ((long)ct * 32 + lane) * 16;

  v8f acc = {};
#pragma unroll
  for (int kt = 0; kt < KT; ++kt) {
    v8bf lo = *(const v8bf*)(arow + kt * 32);        // elements 0..7
    v8bf hi = *(const v8bf*)(arow + kt * 32 + 16);   // elements 8..15
    v16bf a = __builtin_shufflevector(lo, hi, 0, 1, 2, 3, 4, 5, 6, 7,
                                      8, 9, 10, 11, 12, 13, 14, 15);
    v16bf b = *(const v16bf*)(bbase + kt * (NT * 512));
    acc = __builtin_amdgcn_wmma_f32_16x16x32_bf16(false, a, false, b,
                                                  (short)0, acc, false, false);
  }
  // C/D layout: VGPR r -> M = r (lanes 0-15) / M = 8+r (lanes 16-31), N = lane&15
  const int  ncol  = ct * 16 + m;
  const long mbase = (long)blockIdx.x * 16 + khalf * 8;
  const float bz = bias[ncol];
  float* __restrict__ hout = H   + mbase * N + ncol;
  float* __restrict__ aout = Agg + mbase * N + ncol;
#pragma unroll
  for (int r = 0; r < 8; ++r) {
    float hv = acc[r];
    float di = dinv[mbase + r];              // uniform per 16-lane half for each r
    hout[r * N] = hv;
    aout[r * N] = fmaf(di * di, hv, bz);
  }
}

// ---------------------------------------------------------------------------
// Edge scatter: one wave per edge, each lane owns FEATS/32 features.
// Index / dinv loads are wave-uniform (scalarizable); payload is one b128
// (or b64) load + FPL f32 atomics into the L2-resident accumulation buffer.
// ---------------------------------------------------------------------------
template <int FEATS>
__global__ void scatter_edges(const float* __restrict__ h, float* __restrict__ out,
                              const int* __restrict__ src, const int* __restrict__ dst,
                              const float* __restrict__ dinv, int E) {
  constexpr int FPL = FEATS / 32;
  int gid  = blockIdx.x * blockDim.x + threadIdx.x;
  int e    = gid >> 5;
  int lane = gid & 31;
  if (e >= E) return;
  int s = src[e], d = dst[e];
  float norm = dinv[s] * dinv[d];
  const float* __restrict__ hp = h   + (long)s * FEATS + lane * FPL;
  float*       __restrict__ op = out + (long)d * FEATS + lane * FPL;
  float v[FPL];
#pragma unroll
  for (int j = 0; j < FPL; ++j) v[j] = hp[j];
#pragma unroll
  for (int j = 0; j < FPL; ++j) atomicAdd(&op[j], v[j] * norm);
}

// ---------------------------------------------------------------------------
// Launch
// ---------------------------------------------------------------------------
extern "C" void kernel_launch(void* const* d_in, const int* in_sizes, int n_in,
                              void* d_out, int out_size, void* d_ws, size_t ws_size,
                              hipStream_t stream) {
  const float* x  = (const float*)d_in[0];
  const int*   ei = (const int*)d_in[1];     // edge_index, [2, E] flat (int32)
  const float* W1 = (const float*)d_in[2];
  const float* b1 = (const float*)d_in[3];
  const float* W2 = (const float*)d_in[4];
  const float* b2 = (const float*)d_in[5];
  const float* W3 = (const float*)d_in[6];
  const float* b3 = (const float*)d_in[7];
  float* out = (float*)d_out;

  const int E = in_sizes[1] / 2;
  const int n = in_sizes[0] / 128;           // 50000, divisible by 16
  const int* src = ei;
  const int* dst = ei + E;

  auto alignup = [](size_t s) { return (s + 255) & ~(size_t)255; };
  char* ws = (char*)d_ws;
  size_t o = 0;
  float*  dinv = (float*)(ws + o);  o += alignup((size_t)n * 4);
  float*  hbuf = (float*)(ws + o);  o += alignup((size_t)n * 128 * 4);
  float*  agg  = (float*)(ws + o);  o += alignup((size_t)n * 128 * 4);
  __bf16* abf  = (__bf16*)(ws + o); o += alignup((size_t)n * 128 * 2);
  __bf16* wp   = (__bf16*)(ws + o);

  const int rowBlocks  = n / 16;
  const int edgeBlocks = (int)(((long)E * 32 + 255) / 256);
  const long tot128    = (long)n * 128;
  const int packBlocks = (int)((tot128 / 4 + 255) / 256);

  // dinv = rsqrt(1 + indegree)
  fill_ones_f32<<<(n + 255) / 256, 256, 0, stream>>>(dinv, n);
  deg_accum<<<(E + 255) / 256, 256, 0, stream>>>(dst, dinv, E);
  rsqrt_inplace<<<(n + 255) / 256, 256, 0, stream>>>(dinv, n);

  // ---- layer 1: 128 -> 128 ----
  pack_a_bf16<false><<<packBlocks, 256, 0, stream>>>(x, abf, tot128 / 4);
  pack_w_bf16<<<(4 * 8 * 512 + 255) / 256, 256, 0, stream>>>(W1, wp, 128, 128);
  gemm_bf16_wmma<8><<<rowBlocks, 256, 0, stream>>>(abf, wp, hbuf, agg, dinv, b1);
  scatter_edges<128><<<edgeBlocks, 256, 0, stream>>>(hbuf, agg, src, dst, dinv, E);

  // ---- layer 2: 128 -> 128 (relu fused into activation pack) ----
  pack_a_bf16<true><<<packBlocks, 256, 0, stream>>>(agg, abf, tot128 / 4);
  pack_w_bf16<<<(4 * 8 * 512 + 255) / 256, 256, 0, stream>>>(W2, wp, 128, 128);
  gemm_bf16_wmma<8><<<rowBlocks, 256, 0, stream>>>(abf, wp, hbuf, agg, dinv, b2);
  scatter_edges<128><<<edgeBlocks, 256, 0, stream>>>(hbuf, agg, src, dst, dinv, E);

  // ---- layer 3: 128 -> 64, writes d_out ----
  pack_a_bf16<true><<<packBlocks, 256, 0, stream>>>(agg, abf, tot128 / 4);
  pack_w_bf16<<<(4 * 4 * 512 + 255) / 256, 256, 0, stream>>>(W3, wp, 128, 64);
  gemm_bf16_wmma<4><<<rowBlocks, 128, 0, stream>>>(abf, wp, hbuf, out, dinv, b3);
  scatter_edges<64><<<edgeBlocks, 256, 0, stream>>>(hbuf, out, src, dst, dinv, E);
}
